// EncoderLayer_11115375362504
// MI455X (gfx1250) — matrix operand order repaired and checked by reading, not compiled, not run on hardware
//
#include <hip/hip_runtime.h>
#include <hip/hip_bf16.h>

typedef __bf16 bf16;
typedef __attribute__((ext_vector_type(16))) __bf16 v16bf;
typedef __attribute__((ext_vector_type(8)))  float  v8f;

#define B_  4
#define L_  1024
#define K_  48
#define C_  128
#define H_  4
#define HD_ 32

// ---------------- helpers ----------------

__device__ __forceinline__ float geluf(float x) {
  return 0.5f * x * (1.0f + erff(x * 0.70710678118654752f));
}

__device__ __forceinline__ float wsum(float v) {
#pragma unroll
  for (int o = 16; o; o >>= 1) v += __shfl_xor(v, o, 32);
  return v;
}
__device__ __forceinline__ float wmaxr(float v) {
#pragma unroll
  for (int o = 16; o; o >>= 1) v = fmaxf(v, __shfl_xor(v, o, 32));
  return v;
}

// A fragment (16x32 bf16, MxK): lanes 0-15 row=lane K={0..15},
// lanes 16-31 row=lane-16 K={16..31}.  src is row-major with ld lda.
__device__ __forceinline__ v16bf ld_a(const bf16* A, int lda, int m0, int k0, int lane) {
  int r  = m0 + (lane & 15);
  int kb = k0 + ((lane >> 4) << 4);
  const bf16* p = A + r * lda + kb;
  v16bf o;
#pragma unroll
  for (int i = 0; i < 8; ++i) o[i] = p[i];
#pragma unroll
  for (int i = 0; i < 8; ++i) o[8 + i] = p[8 + i];
  return o;
}

// B fragment (32x16 bf16, KxN) where the source W is [N][K] row-major
// (PyTorch-style weight (do,di); B[k][n] = W[n][k]).  Lanes 0-15: n=lane,
// k=k0..k0+15 (contiguous); lanes 16-31: n=lane-16, k=k0+16..k0+31.
__device__ __forceinline__ v16bf ld_b_nk(const bf16* W, int ldw, int n0, int k0, int lane) {
  int n  = n0 + (lane & 15);
  int kb = k0 + ((lane >> 4) << 4);
  const bf16* p = W + n * ldw + kb;
  v16bf o;
#pragma unroll
  for (int i = 0; i < 16; ++i) o[i] = p[i];
  return o;
}

// B fragment where the source M is [K][N] row-major: strided per-lane loads.
__device__ __forceinline__ v16bf ld_b_kn(const bf16* M, int ldm, int k0, int n0, int lane) {
  int n  = n0 + (lane & 15);
  int kb = k0 + ((lane >> 4) << 4);
  v16bf o;
#pragma unroll
  for (int i = 0; i < 16; ++i) o[i] = M[(long)(kb + i) * ldm + n];
  return o;
}

__device__ __forceinline__ v8f wmma_bf(v16bf a, v16bf b, v8f c) {
  return __builtin_amdgcn_wmma_f32_16x16x32_bf16(false, a, false, b, (short)0, c, false, false);
}

// ---------------- weight conversion ----------------

__global__ void cvt_bf16_kernel(const float* __restrict__ s, bf16* __restrict__ d, int n) {
  int i = blockIdx.x * 256 + threadIdx.x;
  if (i < n) d[i] = (bf16)s[i];
}

// ---------------- node message: gather+concat -> MLP -> masked sum -> LN1 ----------------

__global__ __launch_bounds__(128) void node_msg_kernel(
    const float* __restrict__ hV, const float* __restrict__ hE,
    const int* __restrict__ Eidx, const float* __restrict__ maskA,
    const bf16* __restrict__ W1, const float* __restrict__ b1,
    const bf16* __restrict__ W2, const float* __restrict__ b2,
    const bf16* __restrict__ W3, const float* __restrict__ b3,
    const float* __restrict__ ng, const float* __restrict__ nb,
    float* __restrict__ outf, bf16* __restrict__ outb)
{
  const int bl   = blockIdx.x;           // b*L + l
  const int tid  = threadIdx.x;
  const int lane = tid & 31, wave = tid >> 5;
  const int b    = bl >> 10;

  extern __shared__ char smem[];
  bf16*  A   = (bf16*)smem;              // 48 x 392
  bf16*  h1  = A  + 48 * 392;            // 48 x 136
  bf16*  h2  = h1 + 48 * 136;            // 48 x 136
  float* msg = (float*)(h2 + 48 * 136);  // 48 x 128
  __shared__ float hvi[128];
  __shared__ float mA[48];
  __shared__ float red[8];

  if (tid < 128) hvi[tid] = hV[bl * C_ + tid];
  if (tid < 48)  mA[tid]  = maskA[bl * K_ + tid];
  __syncthreads();

  const float* hEr  = hE + (long)bl * K_ * C_;
  const int*   idxr = Eidx + bl * K_;
  for (int t = tid; t < K_ * 384; t += 128) {
    int j = t / 384, c = t - j * 384;
    float v;
    if (c < 128)      v = hvi[c];
    else if (c < 256) v = hEr[j * C_ + (c - 128)];
    else              v = hV[((long)b * L_ + idxr[j]) * C_ + (c - 256)];
    A[j * 392 + c] = (bf16)v;
  }
  __syncthreads();

  // GEMM1: [48x384] x W1(128x384)^T -> h1, +bias, GELU
  for (int nt = wave * 2; nt < wave * 2 + 2; ++nt)
    for (int mt = 0; mt < 3; ++mt) {
      v8f acc = {};
      for (int kt = 0; kt < 12; ++kt)
        acc = wmma_bf(ld_a(A, 392, mt * 16, kt * 32, lane),
                      ld_b_nk(W1, 384, nt * 16, kt * 32, lane), acc);
      int n = nt * 16 + (lane & 15);
      float bias = b1[n];
      int r0 = mt * 16 + ((lane >> 4) << 3);
#pragma unroll
      for (int i = 0; i < 8; ++i) h1[(r0 + i) * 136 + n] = (bf16)geluf(acc[i] + bias);
    }
  __syncthreads();

  // GEMM2
  for (int nt = wave * 2; nt < wave * 2 + 2; ++nt)
    for (int mt = 0; mt < 3; ++mt) {
      v8f acc = {};
      for (int kt = 0; kt < 4; ++kt)
        acc = wmma_bf(ld_a(h1, 136, mt * 16, kt * 32, lane),
                      ld_b_nk(W2, 128, nt * 16, kt * 32, lane), acc);
      int n = nt * 16 + (lane & 15);
      float bias = b2[n];
      int r0 = mt * 16 + ((lane >> 4) << 3);
#pragma unroll
      for (int i = 0; i < 8; ++i) h2[(r0 + i) * 136 + n] = (bf16)geluf(acc[i] + bias);
    }
  __syncthreads();

  // GEMM3 -> msg (fp32, +bias)
  for (int nt = wave * 2; nt < wave * 2 + 2; ++nt)
    for (int mt = 0; mt < 3; ++mt) {
      v8f acc = {};
      for (int kt = 0; kt < 4; ++kt)
        acc = wmma_bf(ld_a(h2, 136, mt * 16, kt * 32, lane),
                      ld_b_nk(W3, 128, nt * 16, kt * 32, lane), acc);
      int n = nt * 16 + (lane & 15);
      float bias = b3[n];
      int r0 = mt * 16 + ((lane >> 4) << 3);
#pragma unroll
      for (int i = 0; i < 8; ++i) msg[(r0 + i) * 128 + n] = acc[i] + bias;
    }
  __syncthreads();

  // masked sum over K, /48, residual, LayerNorm (norm1)
  float s = 0.f;
  for (int j = 0; j < K_; ++j) s += msg[j * 128 + tid] * mA[j];
  float x = hvi[tid] + s * (1.0f / 48.0f);
  float p1 = wsum(x), p2 = wsum(x * x);
  if (lane == 0) { red[wave] = p1; red[4 + wave] = p2; }
  __syncthreads();
  float mu  = (red[0] + red[1] + red[2] + red[3]) * (1.0f / 128.0f);
  float var = (red[4] + red[5] + red[6] + red[7]) * (1.0f / 128.0f) - mu * mu;
  float y = (x - mu) * rsqrtf(var + 1e-5f) * ng[tid] + nb[tid];
  outf[bl * C_ + tid] = y;
  outb[bl * C_ + tid] = (bf16)y;
}

// ---------------- FFN: x -> Win(GELU) -> Wout + residual -> LN2 -> *mask ----------------

__global__ __launch_bounds__(128) void ffn_kernel(
    const bf16* __restrict__ xb, const float* __restrict__ xf,
    const float* __restrict__ maskV,
    const bf16* __restrict__ Win, const float* __restrict__ bin,
    const bf16* __restrict__ Wout, const float* __restrict__ bout,
    const float* __restrict__ ng, const float* __restrict__ nb,
    float* __restrict__ outf, bf16* __restrict__ outb)
{
  const int rb = blockIdx.x * 16;
  const int tid = threadIdx.x, lane = tid & 31, wave = tid >> 5;
  extern __shared__ char smem[];
  bf16*  Ax    = (bf16*)smem;             // 16 x 136
  bf16*  t1    = Ax + 16 * 136;           // 16 x 520
  float* y     = (float*)(t1 + 16 * 520); // 16 x 128
  float* resid = y + 16 * 128;            // 16 x 128

  for (int t = tid; t < 16 * 128; t += 128) {
    int r = t >> 7, c = t & 127;
    Ax[r * 136 + c]    = xb[(rb + r) * C_ + c];
    resid[r * 128 + c] = xf[(rb + r) * C_ + c];
  }
  __syncthreads();

  for (int nt = wave * 8; nt < wave * 8 + 8; ++nt) {
    v8f acc = {};
    for (int kt = 0; kt < 4; ++kt)
      acc = wmma_bf(ld_a(Ax, 136, 0, kt * 32, lane),
                    ld_b_nk(Win, 128, nt * 16, kt * 32, lane), acc);
    int n = nt * 16 + (lane & 15);
    float bias = bin[n];
    int r0 = (lane >> 4) << 3;
#pragma unroll
    for (int i = 0; i < 8; ++i) t1[(r0 + i) * 520 + n] = (bf16)geluf(acc[i] + bias);
  }
  __syncthreads();

  for (int nt = wave * 2; nt < wave * 2 + 2; ++nt) {
    v8f acc = {};
    for (int kt = 0; kt < 16; ++kt)
      acc = wmma_bf(ld_a(t1, 520, 0, kt * 32, lane),
                    ld_b_nk(Wout, 512, nt * 16, kt * 32, lane), acc);
    int n = nt * 16 + (lane & 15);
    float bias = bout[n];
    int r0 = (lane >> 4) << 3;
#pragma unroll
    for (int i = 0; i < 8; ++i)
      y[(r0 + i) * 128 + n] = acc[i] + bias + resid[(r0 + i) * 128 + n];
  }
  __syncthreads();

  for (int r = wave; r < 16; r += 4) {
    int c0 = lane * 4;
    float v[4], s1 = 0.f, s2 = 0.f;
#pragma unroll
    for (int q = 0; q < 4; ++q) { float t = y[r * 128 + c0 + q]; v[q] = t; s1 += t; s2 += t * t; }
    s1 = wsum(s1); s2 = wsum(s2);
    float mu = s1 * (1.0f / 128.0f), var = s2 * (1.0f / 128.0f) - mu * mu;
    float rs = rsqrtf(var + 1e-5f);
    float m = maskV[rb + r];
#pragma unroll
    for (int q = 0; q < 4; ++q) {
      float o = ((v[q] - mu) * rs * ng[c0 + q] + nb[c0 + q]) * m;
      outf[(rb + r) * C_ + c0 + q] = o;
      outb[(rb + r) * C_ + c0 + q] = (bf16)o;
    }
  }
}

// ---------------- edge message: concat -> MLP -> +h_E -> LN3 -> out ----------------

__global__ __launch_bounds__(128) void edge_msg_kernel(
    const bf16* __restrict__ hVb, const float* __restrict__ hE,
    const int* __restrict__ Eidx,
    const bf16* __restrict__ W1, const float* __restrict__ b1,
    const bf16* __restrict__ W2, const float* __restrict__ b2,
    const bf16* __restrict__ W3, const float* __restrict__ b3,
    const float* __restrict__ ng, const float* __restrict__ nb,
    float* __restrict__ outE)
{
  const int bl = blockIdx.x;
  const int tid = threadIdx.x, lane = tid & 31, wave = tid >> 5;
  const int b = bl >> 10;

  extern __shared__ char smem[];
  bf16*  A   = (bf16*)smem;
  bf16*  h1  = A  + 48 * 392;
  bf16*  h2  = h1 + 48 * 136;
  float* msg = (float*)(h2 + 48 * 136);
  __shared__ bf16 hvib[128];

  if (tid < 128) hvib[tid] = hVb[bl * C_ + tid];
  __syncthreads();

  const float* hEr  = hE + (long)bl * K_ * C_;
  const int*   idxr = Eidx + bl * K_;
  for (int t = tid; t < K_ * 384; t += 128) {
    int j = t / 384, c = t - j * 384;
    bf16 v;
    if (c < 128)      v = hvib[c];
    else if (c < 256) v = (bf16)hEr[j * C_ + (c - 128)];
    else              v = hVb[((long)b * L_ + idxr[j]) * C_ + (c - 256)];
    A[j * 392 + c] = v;
  }
  __syncthreads();

  for (int nt = wave * 2; nt < wave * 2 + 2; ++nt)
    for (int mt = 0; mt < 3; ++mt) {
      v8f acc = {};
      for (int kt = 0; kt < 12; ++kt)
        acc = wmma_bf(ld_a(A, 392, mt * 16, kt * 32, lane),
                      ld_b_nk(W1, 384, nt * 16, kt * 32, lane), acc);
      int n = nt * 16 + (lane & 15);
      float bias = b1[n];
      int r0 = mt * 16 + ((lane >> 4) << 3);
#pragma unroll
      for (int i = 0; i < 8; ++i) h1[(r0 + i) * 136 + n] = (bf16)geluf(acc[i] + bias);
    }
  __syncthreads();

  for (int nt = wave * 2; nt < wave * 2 + 2; ++nt)
    for (int mt = 0; mt < 3; ++mt) {
      v8f acc = {};
      for (int kt = 0; kt < 4; ++kt)
        acc = wmma_bf(ld_a(h1, 136, mt * 16, kt * 32, lane),
                      ld_b_nk(W2, 128, nt * 16, kt * 32, lane), acc);
      int n = nt * 16 + (lane & 15);
      float bias = b2[n];
      int r0 = mt * 16 + ((lane >> 4) << 3);
#pragma unroll
      for (int i = 0; i < 8; ++i) h2[(r0 + i) * 136 + n] = (bf16)geluf(acc[i] + bias);
    }
  __syncthreads();

  for (int nt = wave * 2; nt < wave * 2 + 2; ++nt)
    for (int mt = 0; mt < 3; ++mt) {
      v8f acc = {};
      for (int kt = 0; kt < 4; ++kt)
        acc = wmma_bf(ld_a(h2, 136, mt * 16, kt * 32, lane),
                      ld_b_nk(W3, 128, nt * 16, kt * 32, lane), acc);
      int n = nt * 16 + (lane & 15);
      float bias = b3[n];
      int r0 = mt * 16 + ((lane >> 4) << 3);
#pragma unroll
      for (int i = 0; i < 8; ++i) msg[(r0 + i) * 128 + n] = acc[i] + bias;
    }
  __syncthreads();

  // per-row residual + LN (norm3); one wave per row, 12 rows/wave
  for (int j = wave; j < K_; j += 4) {
    int c0 = lane * 4;
    float v[4], s1 = 0.f, s2 = 0.f;
#pragma unroll
    for (int q = 0; q < 4; ++q) {
      float t = hEr[j * C_ + c0 + q] + msg[j * 128 + c0 + q];
      v[q] = t; s1 += t; s2 += t * t;
    }
    s1 = wsum(s1); s2 = wsum(s2);
    float mu = s1 * (1.0f / 128.0f), var = s2 * (1.0f / 128.0f) - mu * mu;
    float rs = rsqrtf(var + 1e-5f);
#pragma unroll
    for (int q = 0; q < 4; ++q)
      outE[((long)bl * K_ + j) * C_ + c0 + q] = (v[q] - mu) * rs * ng[c0 + q] + nb[c0 + q];
  }
}

// ---------------- QKV projection ----------------

__global__ __launch_bounds__(128) void qkv_kernel(
    const bf16* __restrict__ xb, const bf16* __restrict__ W, const float* __restrict__ bw,
    bf16* __restrict__ qkv)
{
  const int rb = blockIdx.x * 16;
  const int tid = threadIdx.x, lane = tid & 31, wave = tid >> 5;
  extern __shared__ char smem[];
  bf16* Ax = (bf16*)smem;  // 16 x 136

  for (int t = tid; t < 16 * 128; t += 128) {
    int r = t >> 7, c = t & 127;
    Ax[r * 136 + c] = xb[(rb + r) * C_ + c];
  }
  __syncthreads();

  for (int nt = wave * 6; nt < wave * 6 + 6; ++nt) {
    v8f acc = {};
    for (int kt = 0; kt < 4; ++kt)
      acc = wmma_bf(ld_a(Ax, 136, 0, kt * 32, lane),
                    ld_b_nk(W, 128, nt * 16, kt * 32, lane), acc);
    int n = nt * 16 + (lane & 15);
    float bias = bw[n];
    int r0 = (lane >> 4) << 3;
#pragma unroll
    for (int i = 0; i < 8; ++i)
      qkv[(long)(rb + r0 + i) * 384 + n] = (bf16)(acc[i] + bias);
  }
}

// ---------------- attention core: scores -> softmax -> P@V ----------------

__global__ __launch_bounds__(128) void attn_kernel(
    const bf16* __restrict__ qkv, const float* __restrict__ maskV, bf16* __restrict__ ob)
{
  const int qt = blockIdx.x;   // 0..63 (16 query rows each)
  const int h  = blockIdx.y;
  const int b  = blockIdx.z;
  const int tid = threadIdx.x, lane = tid & 31, wave = tid >> 5;

  extern __shared__ char smem[];
  float* sc    = (float*)smem;              // 16 x 1024 fp32
  bf16*  P     = (bf16*)(sc + 16 * 1024);   // 16 x 1024 bf16
  bf16*  qtile = P + 16 * 1024;             // 16 x 40
  float* op    = (float*)(qtile + 16 * 40); // 4 x 16 x 32 fp32

  const bf16* qbase = qkv + (long)b * L_ * 384;
  for (int t = tid; t < 512; t += 128) {
    int r = t >> 5, d = t & 31;
    qtile[r * 40 + d] = qbase[(qt * 16 + r) * 384 + h * 32 + d];
  }
  __syncthreads();

  const bf16* kbase = qbase + 128 + h * 32;
  const float scale = 0.17677669529663687f;  // 1/sqrt(32)
  for (int nt = wave; nt < 64; nt += 4) {
    v8f acc = {};
    acc = wmma_bf(ld_a(qtile, 40, 0, 0, lane),
                  ld_b_nk(kbase, 384, nt * 16, 0, lane), acc);
    int n = nt * 16 + (lane & 15);              // key index
    float mk = maskV[b * L_ + n];
    int r0 = (lane >> 4) << 3;
#pragma unroll
    for (int i = 0; i < 8; ++i) {
      float s = acc[i] * scale;
      if (!(mk > 0.f)) s = -1e9f;
      sc[(r0 + i) * 1024 + n] = s;
    }
  }
  __syncthreads();

  // softmax, one wave per row
  for (int r = wave; r < 16; r += 4) {
    float mx = -3.4e38f;
    for (int c = lane; c < 1024; c += 32) mx = fmaxf(mx, sc[r * 1024 + c]);
    mx = wmaxr(mx);
    float s = 0.f;
    for (int c = lane; c < 1024; c += 32) {
      float e = __expf(sc[r * 1024 + c] - mx);
      sc[r * 1024 + c] = e; s += e;
    }
    s = wsum(s);
    float inv = 1.0f / s;
    for (int c = lane; c < 1024; c += 32) P[r * 1024 + c] = (bf16)(sc[r * 1024 + c] * inv);
  }
  __syncthreads();

  // O = P @ V, K split across the 4 waves, partials reduced through LDS
  const bf16* vbase = qbase + 256 + h * 32;
  v8f a0 = {}, a1 = {};
  for (int kt = wave; kt < 32; kt += 4) {
    v16bf ap = ld_a(P, 1024, 0, kt * 32, lane);
    a0 = wmma_bf(ap, ld_b_kn(vbase, 384, kt * 32, 0,  lane), a0);
    a1 = wmma_bf(ap, ld_b_kn(vbase, 384, kt * 32, 16, lane), a1);
  }
  {
    int n = lane & 15;
    int r0 = (lane >> 4) << 3;
#pragma unroll
    for (int i = 0; i < 8; ++i) {
      op[(wave * 16 + r0 + i) * 32 + n]      = a0[i];
      op[(wave * 16 + r0 + i) * 32 + 16 + n] = a1[i];
    }
  }
  __syncthreads();
  for (int t = tid; t < 512; t += 128) {
    int r = t >> 5, d = t & 31;
    float s = op[r * 32 + d] + op[(16 + r) * 32 + d] + op[(32 + r) * 32 + d] + op[(48 + r) * 32 + d];
    ob[(long)(b * L_ + qt * 16 + r) * C_ + h * 32 + d] = (bf16)s;
  }
}

// ---------------- post: out-proj + LN + fc1/fc2 + LN + res_out(GELU) ----------------

__global__ __launch_bounds__(128) void post_kernel(
    const bf16* __restrict__ ob, const float* __restrict__ hv2f,
    const bf16* __restrict__ Wo,  const float* __restrict__ bo,
    const bf16* __restrict__ Wf1, const float* __restrict__ bf1,
    const bf16* __restrict__ Wf2, const float* __restrict__ bf2,
    const bf16* __restrict__ Wr,  const float* __restrict__ br,
    const float* __restrict__ lg, const float* __restrict__ lb,
    float* __restrict__ out)
{
  const int rb = blockIdx.x * 16;
  const int tid = threadIdx.x, lane = tid & 31, wave = tid >> 5;
  extern __shared__ char smem[];
  bf16*  Ao    = (bf16*)smem;              // 16 x 136
  bf16*  tb    = Ao + 16 * 136;            // 16 x 136
  bf16*  t1    = tb + 16 * 136;            // 16 x 264
  bf16*  ub    = t1 + 16 * 264;            // 16 x 136
  float* xx    = (float*)(ub + 16 * 136);  // 16 x 128
  float* resid = xx + 16 * 128;            // 16 x 128

  for (int t = tid; t < 16 * 128; t += 128) {
    int r = t >> 7, c = t & 127;
    Ao[r * 136 + c]    = ob[(rb + r) * C_ + c];
    resid[r * 128 + c] = hv2f[(rb + r) * C_ + c];
  }
  __syncthreads();

  // out-proj + residual
  for (int nt = wave * 2; nt < wave * 2 + 2; ++nt) {
    v8f acc = {};
    for (int kt = 0; kt < 4; ++kt)
      acc = wmma_bf(ld_a(Ao, 136, 0, kt * 32, lane),
                    ld_b_nk(Wo, 128, nt * 16, kt * 32, lane), acc);
    int n = nt * 16 + (lane & 15);
    float bias = bo[n];
    int r0 = (lane >> 4) << 3;
#pragma unroll
    for (int i = 0; i < 8; ++i)
      xx[(r0 + i) * 128 + n] = acc[i] + bias + resid[(r0 + i) * 128 + n];
  }
  __syncthreads();

  // LN (ln_attn) -> resid (res2, fp32) + tb (bf16)
  for (int r = wave; r < 16; r += 4) {
    int c0 = lane * 4;
    float v[4], s1 = 0.f, s2 = 0.f;
#pragma unroll
    for (int q = 0; q < 4; ++q) { float t = xx[r * 128 + c0 + q]; v[q] = t; s1 += t; s2 += t * t; }
    s1 = wsum(s1); s2 = wsum(s2);
    float mu = s1 * (1.0f / 128.0f), var = s2 * (1.0f / 128.0f) - mu * mu;
    float rs = rsqrtf(var + 1e-5f);
#pragma unroll
    for (int q = 0; q < 4; ++q) {
      float t = (v[q] - mu) * rs * lg[c0 + q] + lb[c0 + q];
      resid[r * 128 + c0 + q] = t;
      tb[r * 136 + c0 + q] = (bf16)t;
    }
  }
  __syncthreads();

  // fc1 (+GELU)
  for (int nt = wave * 4; nt < wave * 4 + 4; ++nt) {
    v8f acc = {};
    for (int kt = 0; kt < 4; ++kt)
      acc = wmma_bf(ld_a(tb, 136, 0, kt * 32, lane),
                    ld_b_nk(Wf1, 128, nt * 16, kt * 32, lane), acc);
    int n = nt * 16 + (lane & 15);
    float bias = bf1[n];
    int r0 = (lane >> 4) << 3;
#pragma unroll
    for (int i = 0; i < 8; ++i) t1[(r0 + i) * 264 + n] = (bf16)geluf(acc[i] + bias);
  }
  __syncthreads();

  // fc2 + residual
  for (int nt = wave * 2; nt < wave * 2 + 2; ++nt) {
    v8f acc = {};
    for (int kt = 0; kt < 8; ++kt)
      acc = wmma_bf(ld_a(t1, 264, 0, kt * 32, lane),
                    ld_b_nk(Wf2, 256, nt * 16, kt * 32, lane), acc);
    int n = nt * 16 + (lane & 15);
    float bias = bf2[n];
    int r0 = (lane >> 4) << 3;
#pragma unroll
    for (int i = 0; i < 8; ++i)
      xx[(r0 + i) * 128 + n] = acc[i] + bias + resid[(r0 + i) * 128 + n];
  }
  __syncthreads();

  // LN (ln_attn again) -> ub
  for (int r = wave; r < 16; r += 4) {
    int c0 = lane * 4;
    float v[4], s1 = 0.f, s2 = 0.f;
#pragma unroll
    for (int q = 0; q < 4; ++q) { float t = xx[r * 128 + c0 + q]; v[q] = t; s1 += t; s2 += t * t; }
    s1 = wsum(s1); s2 = wsum(s2);
    float mu = s1 * (1.0f / 128.0f), var = s2 * (1.0f / 128.0f) - mu * mu;
    float rs = rsqrtf(var + 1e-5f);
#pragma unroll
    for (int q = 0; q < 4; ++q)
      ub[r * 136 + c0 + q] = (bf16)((v[q] - mu) * rs * lg[c0 + q] + lb[c0 + q]);
  }
  __syncthreads();

  // res_out + GELU -> final node output
  for (int nt = wave * 2; nt < wave * 2 + 2; ++nt) {
    v8f acc = {};
    for (int kt = 0; kt < 4; ++kt)
      acc = wmma_bf(ld_a(ub, 136, 0, kt * 32, lane),
                    ld_b_nk(Wr, 128, nt * 16, kt * 32, lane), acc);
    int n = nt * 16 + (lane & 15);
    float bias = br[n];
    int r0 = (lane >> 4) << 3;
#pragma unroll
    for (int i = 0; i < 8; ++i)
      out[(rb + r0 + i) * C_ + n] = geluf(acc[i] + bias);
  }
}

// ---------------- host launch ----------------

extern "C" void kernel_launch(void* const* d_in, const int* in_sizes, int n_in,
                              void* d_out, int out_size, void* d_ws, size_t ws_size,
                              hipStream_t stream) {
  (void)in_sizes; (void)n_in; (void)out_size; (void)ws_size;

  const float* hV    = (const float*)d_in[0];
  const float* hE    = (const float*)d_in[1];
  const int*   Eidx  = (const int*)d_in[2];
  const float* maskV = (const float*)d_in[3];
  const float* maskA = (const float*)d_in[4];
  const float* W1w  = (const float*)d_in[5];  const float* W1b  = (const float*)d_in[6];
  const float* W2w  = (const float*)d_in[7];  const float* W2b  = (const float*)d_in[8];
  const float* W3w  = (const float*)d_in[9];  const float* W3b  = (const float*)d_in[10];
  const float* W11w = (const float*)d_in[11]; const float* W11b = (const float*)d_in[12];
  const float* W12w = (const float*)d_in[13]; const float* W12b = (const float*)d_in[14];
  const float* W13w = (const float*)d_in[15]; const float* W13b = (const float*)d_in[16];
  const float* Winw = (const float*)d_in[17]; const float* Winb = (const float*)d_in[18];
  const float* Woutw= (const float*)d_in[19]; const float* Woutb= (const float*)d_in[20];
  const float* Aiw  = (const float*)d_in[21]; const float* Aib  = (const float*)d_in[22];
  const float* Aow  = (const float*)d_in[23]; const float* Aob  = (const float*)d_in[24];
  const float* F1w  = (const float*)d_in[25]; const float* F1b  = (const float*)d_in[26];
  const float* F2w  = (const float*)d_in[27]; const float* F2b  = (const float*)d_in[28];
  const float* Rw   = (const float*)d_in[29]; const float* Rb   = (const float*)d_in[30];
  const float* n1g  = (const float*)d_in[31]; const float* n1b  = (const float*)d_in[32];
  const float* n2g  = (const float*)d_in[33]; const float* n2b  = (const float*)d_in[34];
  const float* n3g  = (const float*)d_in[35]; const float* n3b  = (const float*)d_in[36];
  const float* lag  = (const float*)d_in[37]; const float* lab  = (const float*)d_in[38];

  char* ws = (char*)d_ws;
  bf16* wb = (bf16*)ws;
  size_t off = 0;
  bf16* W1f  = wb + off; off += 128 * 384;
  bf16* W2f  = wb + off; off += 128 * 128;
  bf16* W3f  = wb + off; off += 128 * 128;
  bf16* W11f = wb + off; off += 128 * 384;
  bf16* W12f = wb + off; off += 128 * 128;
  bf16* W13f = wb + off; off += 128 * 128;
  bf16* Winf = wb + off; off += 512 * 128;
  bf16* Woutf= wb + off; off += 128 * 512;
  bf16* Aif  = wb + off; off += 384 * 128;
  bf16* Aof  = wb + off; off += 128 * 128;
  bf16* F1f  = wb + off; off += 256 * 128;
  bf16* F2f  = wb + off; off += 128 * 256;
  bf16* Rf   = wb + off; off += 128 * 128;

  float* hV1f = (float*)(ws + (size_t)( 1 << 20));
  bf16*  hV1b = (bf16*) (ws + (size_t)( 3 << 20));
  float* hV2f = (float*)(ws + (size_t)( 4 << 20));
  bf16*  hV2b = (bf16*) (ws + (size_t)( 6 << 20));
  bf16*  qkvb = (bf16*) (ws + (size_t)( 7 << 20));
  bf16*  obuf = (bf16*) (ws + (size_t)(10 << 20));

  struct CvtJob { const float* s; bf16* d; int n; };
  CvtJob cv[13] = {
    {W1w,  W1f,  128 * 384}, {W2w,  W2f,  128 * 128}, {W3w,  W3f,  128 * 128},
    {W11w, W11f, 128 * 384}, {W12w, W12f, 128 * 128}, {W13w, W13f, 128 * 128},
    {Winw, Winf, 512 * 128}, {Woutw,Woutf,128 * 512}, {Aiw,  Aif,  384 * 128},
    {Aow,  Aof,  128 * 128}, {F1w,  F1f,  256 * 128}, {F2w,  F2f,  128 * 256},
    {Rw,   Rf,   128 * 128}
  };
  for (int i = 0; i < 13; ++i)
    cvt_bf16_kernel<<<(cv[i].n + 255) / 256, 256, 0, stream>>>(cv[i].s, cv[i].d, cv[i].n);

  const size_t smMsg  = (size_t)(48 * 392 + 2 * 48 * 136) * 2 + (size_t)48 * 128 * 4;
  const size_t smFfn  = (size_t)(16 * 136 + 16 * 520) * 2 + (size_t)2 * 16 * 128 * 4;
  const size_t smQkv  = (size_t)16 * 136 * 2;
  const size_t smAttn = (size_t)16 * 1024 * 4 + (size_t)(16 * 1024 + 16 * 40) * 2 + (size_t)4 * 16 * 32 * 4;
  const size_t smPost = (size_t)(3 * 16 * 136 + 16 * 264) * 2 + (size_t)2 * 16 * 128 * 4;

  node_msg_kernel<<<B_ * L_, 128, smMsg, stream>>>(
      hV, hE, Eidx, maskA, W1f, W1b, W2f, W2b, W3f, W3b, n1g, n1b, hV1f, hV1b);

  ffn_kernel<<<(B_ * L_) / 16, 128, smFfn, stream>>>(
      hV1b, hV1f, maskV, Winf, Winb, Woutf, Woutb, n2g, n2b, hV2f, hV2b);

  edge_msg_kernel<<<B_ * L_, 128, smMsg, stream>>>(
      hV2b, hE, Eidx, W11f, W11b, W12f, W12b, W13f, W13b, n3g, n3b,
      (float*)d_out + (size_t)B_ * L_ * C_);

  qkv_kernel<<<(B_ * L_) / 16, 128, smQkv, stream>>>(hV2b, Aif, Aib, qkvb);

  attn_kernel<<<dim3(L_ / 16, H_, B_), 128, smAttn, stream>>>(qkvb, maskV, obuf);

  post_kernel<<<(B_ * L_) / 16, 128, smPost, stream>>>(
      obuf, hV2f, Aof, Aob, F1f, F1b, F2f, F2b, Rf, Rb, lag, lab, (float*)d_out);
}
